// Y_Model_40922448396717
// MI455X (gfx1250) — matrix operand order repaired and checked by reading, not compiled, not run on hardware
//
#include <hip/hip_runtime.h>
#include <math.h>

// ---------------------------------------------------------------------------
// Fused DLRM-style attention model for MI455X (gfx1250, wave32, WMMA).
// One workgroup (256 threads = 8 wave32) per batch element; all intermediates
// ([F,H] Xn/Q/K f16, per-wave [16,F] score strips f32) live in LDS so the
// 507MB [B,F,F] attention tensor never touches HBM. GEMMs use
// v_wmma_f32_16x16x32_f16 (f16 in, f32 accumulate). attn@V is dead code in
// the reference and is skipped. Fragment traffic uses 16-byte LDS vectors
// (ds_load_b128) and Qw/Kw are staged once per WG as f16 in LDS (aliased
// over the phase-E strip region).
// ---------------------------------------------------------------------------

#define BATCH      1024
#define XD         64
#define XSD        32
#define SED        8
#define SSEQ       32
#define SDIM       16
#define HD         64      // hidden H
#define FD         352     // feature count F = 22*16
#define FC1_IN     384
#define NTHREADS   256
#define NWAVES     8
#define RT_TILES   22      // F/16
#define QK_TASKS   176     // 2 matrices * 22 row tiles * 4 col tiles

typedef _Float16 v16h __attribute__((ext_vector_type(16)));
typedef _Float16 v8h  __attribute__((ext_vector_type(8)));
typedef float    v8f  __attribute__((ext_vector_type(8)));

// ---- dynamic LDS layout (bytes) -------------------------------------------
#define OFF_XC     0                    // 352 f32   (Xc)
#define OFF_MLPIN  1408                 // 384 f32   (X_attn | sx | rx)
#define OFF_RNORM  2944                 // 64  f32
#define OFF_RED    3200                 // 4*64 f32  (norm partials)
#define OFF_HX     4224                 // 2*64 f32  (sigmoid(Xc@w2^T+b))
#define OFF_ISC    4736                 // 2*16 f32  (interaction scores)
#define OFF_M1     4864                 // 64 f32
#define OFF_M2     5120                 // 64 f32
#define OFF_XN     5376                 // 352*64 f16 = 45056
#define OFF_QS     50432                // 352*64 f16
#define OFF_KS     95488                // 352*64 f16
#define OFF_STRIP  140544               // 8 waves * 16*352 f32 = 180224
// f16 staged weights alias the strip region (live only in phase D):
#define OFF_QWF    OFF_STRIP            // 64*64 f16 = 8192
#define OFF_KWF    (OFF_STRIP + 8192)   // 64*64 f16 = 8192
#define SMEM_BYTES 320768               // <= 320KB per WGP

// ---- WMMA fragment loaders (ISA 7.12.2 layouts, b128 LDS reads) -----------
// A 16x32 f16: lane<16 -> M=lane,    elems 0..7 = K0..7,  8..15 = K16..23
//              lane>=16 -> M=lane-16, elems 0..7 = K8..15, 8..15 = K24..31
// Each half of the fragment is 8 contiguous halves = one 16B LDS load.
__device__ __forceinline__ v16h frag_a_lds(const _Float16* p, int row0, int k0, int ld) {
    int lane = threadIdx.x & 31;
    int m  = row0 + (lane & 15);
    int kb = k0 + ((lane >> 4) << 3);
    const _Float16* base = p + m * ld + kb;
    v8h lo = *(const v8h*)(base);        // K kb .. kb+7
    v8h hi = *(const v8h*)(base + 16);   // K kb+16 .. kb+23
    return __builtin_shufflevector(lo, hi, 0, 1, 2, 3, 4, 5, 6, 7,
                                           8, 9, 10, 11, 12, 13, 14, 15);
}

// B 32x16 f16 where B[d][n] = Wf16[(col0+n)*ld + k0+d] (transposed feed).
// lane<16 -> N=lane, elems = K0..7,K16..23 ; lane>=16 -> N=lane-16, elems = K8..15,K24..31
// two runs of 8 contiguous halves = two 16B LDS loads.
__device__ __forceinline__ v16h frag_b_ldsT(const _Float16* p, int col0, int k0, int ld) {
    int lane = threadIdx.x & 31;
    int n  = lane & 15;
    int db = k0 + ((lane >> 4) << 3);
    const _Float16* base = p + (col0 + n) * ld + db;
    v8h lo = *(const v8h*)(base);
    v8h hi = *(const v8h*)(base + 16);
    return __builtin_shufflevector(lo, hi, 0, 1, 2, 3, 4, 5, 6, 7,
                                           8, 9, 10, 11, 12, 13, 14, 15);
}

__device__ __forceinline__ float sigmoidf_fast(float x) {
    return 1.0f / (1.0f + __expf(-x));
}

__global__ void __launch_bounds__(NTHREADS)
ymodel_fused_kernel(const float* __restrict__ X, const float* __restrict__ Xsp,
                    const float* __restrict__ Xseq, const float* __restrict__ S,
                    const float* __restrict__ R, const float* __restrict__ emb,
                    const float* __restrict__ xrw,
                    const float* __restrict__ Qw, const float* __restrict__ Qb,
                    const float* __restrict__ Kw, const float* __restrict__ Kb,
                    const float* __restrict__ ws1, const float* __restrict__ bs1,
                    const float* __restrict__ wx2, const float* __restrict__ bx2,
                    const float* __restrict__ wsx33, const float* __restrict__ bsx33,
                    const float* __restrict__ wr1, const float* __restrict__ br1,
                    const float* __restrict__ wxx2, const float* __restrict__ bxx2,
                    const float* __restrict__ wrx33, const float* __restrict__ brx33,
                    const float* __restrict__ fc1w, const float* __restrict__ fc1b,
                    const float* __restrict__ fc2w, const float* __restrict__ fc2b,
                    const float* __restrict__ fc3w, const float* __restrict__ fc3b,
                    float* __restrict__ out) {
    extern __shared__ char smem[];
    float*     xc    = (float*)(smem + OFF_XC);
    float*     mlpin = (float*)(smem + OFF_MLPIN);   // [0,352)=X_attn, 352..=sx,rx
    float*     xattn = mlpin;
    float*     rnorm = (float*)(smem + OFF_RNORM);
    float*     red   = (float*)(smem + OFF_RED);
    float*     hx    = (float*)(smem + OFF_HX);
    float*     isc   = (float*)(smem + OFF_ISC);
    float*     m1    = (float*)(smem + OFF_M1);
    float*     m2    = (float*)(smem + OFF_M2);
    _Float16*  xn    = (_Float16*)(smem + OFF_XN);
    _Float16*  qs    = (_Float16*)(smem + OFF_QS);
    _Float16*  ks    = (_Float16*)(smem + OFF_KS);
    _Float16*  qwf   = (_Float16*)(smem + OFF_QWF);
    _Float16*  kwf   = (_Float16*)(smem + OFF_KWF);

    const int b    = blockIdx.x;
    const int tid  = threadIdx.x;
    const int wave = tid >> 5;
    const int lane = tid & 31;

    // -------- Phase A: Xc = concat(X, outer(X_sparse, emb).flatten, X_seq) --
    for (int i = tid; i < FD; i += NTHREADS) {
        float v;
        if (i < XD) {
            v = X[b * XD + i];
        } else if (i < XD + XSD * SED) {
            int j = i - XD;                   // j = sparse_idx*8 + emb_idx
            v = Xsp[b * XSD + (j >> 3)] * emb[j];
        } else {
            v = Xseq[b * SSEQ + (i - XD - XSD * SED)];
        }
        xc[i]    = v;
        xattn[i] = 0.0f;
    }
    // Stage Qw/Kw as f16 into LDS (8KB each), 8 elems per thread per iter.
    for (int i = tid * 8; i < HD * HD; i += NTHREADS * 8) {
        v8h qa, ka;
#pragma unroll
        for (int j = 0; j < 8; ++j) {
            qa[j] = (_Float16)Qw[i + j];
            ka[j] = (_Float16)Kw[i + j];
        }
        *(v8h*)(qwf + i) = qa;
        *(v8h*)(kwf + i) = ka;
    }
    __syncthreads();

    // -------- Phase B: per-column norms over F (axis=1 of [B,F,H]) ---------
    {
        int d = tid & 63, part = tid >> 6;              // 4 partials per column
        float s = 0.0f;
        int f0 = part * (FD / 4), f1 = f0 + (FD / 4);
        for (int f = f0; f < f1; ++f) {
            float v = xc[f] * xrw[f * HD + d];
            s += v * v;
        }
        red[part * 64 + d] = s;
    }
    __syncthreads();
    if (tid < 64) {
        float s = red[tid] + red[64 + tid] + red[128 + tid] + red[192 + tid];
        rnorm[tid] = rsqrtf(s);
    }
    __syncthreads();

    // -------- Phase C: Xn (f16) = Xc[f]*W[f,d]*rnorm[d], 8-wide ------------
    for (int i0 = tid * 8; i0 < FD * HD; i0 += NTHREADS * 8) {
        int f = i0 >> 6, d = i0 & 63;
        float xcv = xc[f];
        v8h t;
#pragma unroll
        for (int j = 0; j < 8; ++j)
            t[j] = (_Float16)(xcv * xrw[i0 + j] * rnorm[d + j]);
        *(v8h*)(xn + i0) = t;
    }
    __syncthreads();

    // -------- Phase D: Q = Xn@Qw^T + Qb ; K = Xn@Kw^T + Kb (WMMA) ----------
    for (int task = wave; task < QK_TASKS; task += NWAVES) {   // wave-uniform
        int mat  = (task >= QK_TASKS / 2);
        int t    = task - mat * (QK_TASKS / 2);
        int row0 = (t >> 2) << 4;
        int col0 = (t & 3) << 4;
        const _Float16* wf   = mat ? kwf : qwf;
        const float*    bias = mat ? Kb : Qb;
        _Float16*       dst  = mat ? ks : qs;
        v8f acc = {};
#pragma unroll
        for (int k0 = 0; k0 < HD; k0 += 32) {
            v16h a  = frag_a_lds(xn, row0, k0, HD);
            v16h bf = frag_b_ldsT(wf, col0, k0, HD);
            acc = __builtin_amdgcn_wmma_f32_16x16x32_f16(
                false, a, false, bf, (short)0, acc, false, false);
        }
        int n = lane & 15, mb = (lane >> 4) << 3;
        float bv = bias[col0 + n];
#pragma unroll
        for (int r = 0; r < 8; ++r)
            dst[(row0 + r + mb) * HD + col0 + n] = (_Float16)(acc[r] + bv);
    }
    __syncthreads();   // qs/ks complete; qwf/kwf dead -> strips may reuse LDS

    // -------- Phase E: per-wave row strips: scores, softmax, X_attn --------
    float* strip = (float*)(smem + OFF_STRIP) + wave * 16 * FD;
    for (int rt = wave; rt < RT_TILES; rt += NWAVES) {         // wave-uniform
        int row0 = rt << 4;
        for (int ct = 0; ct < RT_TILES; ++ct) {
            int col0 = ct << 4;
            v8f acc = {};
#pragma unroll
            for (int k0 = 0; k0 < HD; k0 += 32) {
                v16h a  = frag_a_lds(qs, row0, k0, HD);
                v16h bf = frag_b_ldsT(ks, col0, k0, HD);
                acc = __builtin_amdgcn_wmma_f32_16x16x32_f16(
                    false, a, false, bf, (short)0, acc, false, false);
            }
            int n = lane & 15, mb = (lane >> 4) << 3;
#pragma unroll
            for (int r = 0; r < 8; ++r)
                strip[(r + mb) * FD + col0 + n] = acc[r] * 0.125f;  // /sqrt(H)
        }
        // row softmax + weighted accumulation into X_attn
        for (int m = 0; m < 16; ++m) {
            float* row = strip + m * FD;
            float mx = -3.4e38f;
            for (int g = lane; g < FD; g += 32) mx = fmaxf(mx, row[g]);
#pragma unroll
            for (int o = 16; o > 0; o >>= 1) mx = fmaxf(mx, __shfl_xor(mx, o, 32));
            float se = 0.0f;
            for (int g = lane; g < FD; g += 32) {
                float e = __expf(row[g] - mx);
                row[g] = e;
                se += e;
            }
#pragma unroll
            for (int o = 16; o > 0; o >>= 1) se += __shfl_xor(se, o, 32);
            float sc = xc[row0 + m] / se;      // Xc[f] * attn[f,g] contribution
            for (int g = lane; g < FD; g += 32) atomicAdd(&xattn[g], row[g] * sc);
        }
    }
    __syncthreads();

    // -------- Phase F: interaction attentions (S and R paths) --------------
    if (tid < 128) {                       // h_x = sigmoid(Xc @ w2^T + b2)
        int which = tid >> 6, h = tid & 63;
        const float* w2 = which ? wxx2 : wx2;
        const float* b2 = which ? bxx2 : bx2;
        float s = b2[h];
        const float* wrow = w2 + h * FD;
        for (int f = 0; f < FD; ++f) s += xc[f] * wrow[f];
        hx[which * 64 + h] = sigmoidf_fast(s);
    }
    __syncthreads();
    if (tid < 32) {                        // per-feature interaction scores
        int which = tid >> 4, i = tid & 15;
        const float* sv  = which ? R : S;
        const float* w1  = which ? wr1 : ws1;
        const float* b1  = which ? br1 : bs1;
        const float* w33 = which ? wrx33 : wsx33;
        float b33 = which ? brx33[0] : bsx33[0];
        float si = sv[b * SDIM + i];
        float sc = b33;
        const float* hxw = hx + which * 64;
        for (int h = 0; h < HD; ++h) {
            float hs = sigmoidf_fast(si * w1[h] + b1[h]);
            sc += fmaxf(hs + hxw[h], 0.0f) * w33[h];
        }
        isc[which * 16 + i] = sc;
    }
    __syncthreads();
    if (tid < 2) {                         // softmax over 16 + gate
        int which = tid;
        const float* sv = which ? R : S;
        const float* sb = isc + which * 16;
        float mx = -3.4e38f;
        for (int i = 0; i < 16; ++i) mx = fmaxf(mx, sb[i]);
        float e[16], se = 0.0f;
        for (int i = 0; i < 16; ++i) { e[i] = __expf(sb[i] - mx); se += e[i]; }
        float inv = 1.0f / se;
        for (int i = 0; i < 16; ++i)
            mlpin[FD + which * 16 + i] = sv[b * SDIM + i] * e[i] * inv;
    }
    __syncthreads();

    // -------- Phase G: MLP head [384]->64->64->1 ---------------------------
    if (tid < 64) {
        float s = fc1b[tid];
        const float* wrow = fc1w + tid * FC1_IN;
        for (int j = 0; j < FC1_IN; ++j) s += mlpin[j] * wrow[j];
        m1[tid] = fmaxf(s, 0.0f);
    }
    __syncthreads();
    if (tid < 64) {
        float s = fc2b[tid];
        const float* wrow = fc2w + tid * HD;
        for (int j = 0; j < HD; ++j) s += m1[j] * wrow[j];
        m2[tid] = fmaxf(s, 0.0f);
    }
    __syncthreads();
    if (tid == 0) {
        float s = fc3b[0];
        for (int j = 0; j < HD; ++j) s += m2[j] * fc3w[j];
        out[b] = s;
    }
}

extern "C" void kernel_launch(void* const* d_in, const int* in_sizes, int n_in,
                              void* d_out, int out_size, void* d_ws, size_t ws_size,
                              hipStream_t stream) {
    (void)in_sizes; (void)n_in; (void)out_size; (void)d_ws; (void)ws_size;
    const float* X     = (const float*)d_in[0];
    const float* Xsp   = (const float*)d_in[1];
    const float* Xseq  = (const float*)d_in[2];
    const float* S     = (const float*)d_in[3];
    const float* R     = (const float*)d_in[4];
    const float* emb   = (const float*)d_in[5];
    const float* xrw   = (const float*)d_in[6];
    const float* Qw    = (const float*)d_in[7];
    const float* Qb    = (const float*)d_in[8];
    const float* Kw    = (const float*)d_in[9];
    const float* Kb    = (const float*)d_in[10];
    // d_in[11]=Vw, d_in[12]=Vb : dead in the reference (attn@V is discarded)
    const float* ws1   = (const float*)d_in[13];
    const float* bs1   = (const float*)d_in[14];
    const float* wx2   = (const float*)d_in[15];
    const float* bx2   = (const float*)d_in[16];
    const float* wsx33 = (const float*)d_in[17];
    const float* bsx33 = (const float*)d_in[18];
    const float* wr1   = (const float*)d_in[19];
    const float* br1   = (const float*)d_in[20];
    const float* wxx2  = (const float*)d_in[21];
    const float* bxx2  = (const float*)d_in[22];
    const float* wrx33 = (const float*)d_in[23];
    const float* brx33 = (const float*)d_in[24];
    const float* fc1w  = (const float*)d_in[25];
    const float* fc1b  = (const float*)d_in[26];
    const float* fc2w  = (const float*)d_in[27];
    const float* fc2b  = (const float*)d_in[28];
    const float* fc3w  = (const float*)d_in[29];
    const float* fc3b  = (const float*)d_in[30];

    (void)hipFuncSetAttribute((const void*)ymodel_fused_kernel,
                              hipFuncAttributeMaxDynamicSharedMemorySize,
                              SMEM_BYTES);

    ymodel_fused_kernel<<<BATCH, NTHREADS, SMEM_BYTES, stream>>>(
        X, Xsp, Xseq, S, R, emb, xrw, Qw, Qb, Kw, Kb,
        ws1, bs1, wx2, bx2, wsx33, bsx33,
        wr1, br1, wxx2, bxx2, wrx33, brx33,
        fc1w, fc1b, fc2w, fc2b, fc3w, fc3b,
        (float*)d_out);
}
